// CLIPModelShuffleAttentionPenultimateLayer_59837484368493
// MI455X (gfx1250) — compile-verified, matrix operand order checked by reading, not run on hardware
//
#include <hip/hip_runtime.h>
#include <stdint.h>

// ---------------------------------------------------------------------------
// CLIPModelShuffleAttentionPenultimateLayer for MI455X (gfx1250, wave32)
//
// Math simplifications (exact):
//   encode_image: mean over tokens commutes with patch-embed GEMM ->
//     f = LN( mean_t(patches) @ W_pe + mean_t(pos_emb) )
//   noise branch: npool = mean4x4(x) - mean8x8(x) (aligned blocks)
//
// Dataflow: producers emit f16 activations (K padded to mult of 32);
// weights are converted once per launch to f16 *pre-transposed* (Bt[n][k]).
// The WMMA GEMM is then guard-free: 16B async copies global->LDS
// (global_load_async_to_lds_b128 + s_wait_asynccnt), ds_load_b128 fragment
// loads, v_wmma_f32_16x16x32_f16 accumulation.
// ---------------------------------------------------------------------------

typedef _Float16 f16x8  __attribute__((ext_vector_type(8)));
typedef _Float16 f16x16 __attribute__((ext_vector_type(16)));
typedef float    f32x8  __attribute__((ext_vector_type(8)));
typedef int      i32x4  __attribute__((ext_vector_type(4)));

#define BATCH  64
#define CH     3
#define HW     224
#define PS     32
#define NPATCH 49
#define CPATCH 14
#define NTOK   256
#define PDIM   588
#define PKPE   608     // PDIM padded to multiple of 32 (19*32)
#define NOISED 9408    // already a multiple of 32
#define DDIM   1024

// --------- CDNA5 async global->LDS copy (guarded; fallback = sync copy) ----
#if defined(__HIP_DEVICE_COMPILE__) && defined(__gfx1250__) && defined(__has_builtin)
#  if __has_builtin(__builtin_amdgcn_global_load_async_to_lds_b128)
#    define HAVE_ASYNC_LDS 1
#  endif
#endif

// builtin signature (from hipcc diagnostic): (i32x4 AS1*, i32x4 AS3*, Imm, Imm)
typedef __attribute__((address_space(1))) i32x4* as1_i4p;
typedef __attribute__((address_space(3))) i32x4* as3_i4p;

__device__ __forceinline__ void copy16B_to_lds(const _Float16* gsrc, _Float16* ldst) {
#ifdef HAVE_ASYNC_LDS
  __builtin_amdgcn_global_load_async_to_lds_b128(
      (as1_i4p)(uintptr_t)gsrc, (as3_i4p)(uintptr_t)ldst, 0, 0);
#else
  *reinterpret_cast<f16x8*>(ldst) = *reinterpret_cast<const f16x8*>(gsrc);
#endif
}

__device__ __forceinline__ void wait_lds_copies() {
#ifdef HAVE_ASYNC_LDS
#  if __has_builtin(__builtin_amdgcn_s_wait_asynccnt)
  __builtin_amdgcn_s_wait_asynccnt((short)0);
#  else
  asm volatile("s_wait_asynccnt 0x0" ::: "memory");
#  endif
#endif
}

// ---------------------------------------------------------------------------
// Producers (emit f16 activations, zero-padded K)
// ---------------------------------------------------------------------------

// r[b, col] = mean over 16x16 tokens of x[b, c, th*14+i, tw*14+j]; col=(c,i,j)
__global__ void k_token_mean(const float* __restrict__ x, _Float16* __restrict__ r) {
  int idx = blockIdx.x * blockDim.x + threadIdx.x;
  if (idx >= BATCH * PKPE) return;
  int b = idx / PKPE, col = idx % PKPE;
  if (col >= PDIM) { r[idx] = (_Float16)0.f; return; }
  int c = col / (CPATCH * CPATCH);
  int ij = col % (CPATCH * CPATCH);
  int i = ij / CPATCH, j = ij % CPATCH;
  const float* xb = x + (size_t)(b * CH + c) * HW * HW;
  float s = 0.f;
  for (int th = 0; th < 16; ++th)
    for (int tw = 0; tw < 16; ++tw)
      s += xb[(th * CPATCH + i) * HW + (tw * CPATCH + j)];
  r[idx] = (_Float16)(s * (1.0f / 256.0f));
}

// Same through the patch shuffle + rot90 + flips (index gather, no xs buffer)
__global__ void k_token_mean_shuf(const float* __restrict__ x,
                                  const int* __restrict__ sidx,
                                  const int* __restrict__ rotk,
                                  const int* __restrict__ fliph,
                                  const int* __restrict__ flipw,
                                  _Float16* __restrict__ r) {
  int idx = blockIdx.x * blockDim.x + threadIdx.x;
  if (idx >= BATCH * PKPE) return;
  int b = idx / PKPE, col = idx % PKPE;
  if (col >= PDIM) { r[idx] = (_Float16)0.f; return; }
  int c = col / (CPATCH * CPATCH);
  int ij = col % (CPATCH * CPATCH);
  int i = ij / CPATCH, j = ij % CPATCH;
  const float* xb = x + (size_t)(b * CH + c) * HW * HW;
  float s = 0.f;
  for (int th = 0; th < 16; ++th) {
    for (int tw = 0; tw < 16; ++tw) {
      int gy = th * CPATCH + i, gx = tw * CPATCH + j;
      int pn = (gy >> 5) * 7 + (gx >> 5);
      int u = gy & 31, v = gx & 31;
      if (fliph[b * NPATCH + pn]) u = 31 - u;
      if (flipw[b * NPATCH + pn]) v = 31 - v;
      int kr = rotk[b * NPATCH + pn] & 3;
      int su, sv;
      if      (kr == 0) { su = u;      sv = v;      }
      else if (kr == 1) { su = v;      sv = 31 - u; }
      else if (kr == 2) { su = 31 - u; sv = 31 - v; }
      else              { su = 31 - v; sv = u;      }
      int sp = sidx[pn];
      int sy = (sp / 7) * PS + su, sx = (sp % 7) * PS + sv;
      s += xb[sy * HW + sx];
    }
  }
  r[idx] = (_Float16)(s * (1.0f / 256.0f));
}

// npool[b, c, yy, xx] = mean4x4(x) - mean8x8(x)
__global__ void k_npool(const float* __restrict__ x, _Float16* __restrict__ np) {
  int idx = blockIdx.x * blockDim.x + threadIdx.x;
  if (idx >= BATCH * NOISED) return;
  int b = idx / NOISED, rem = idx % NOISED;
  int c = rem / (56 * 56), yx = rem % (56 * 56);
  int yy = yx / 56, xx = yx % 56;
  const float* xb = x + (size_t)(b * CH + c) * HW * HW;
  float s4 = 0.f;
  for (int dy = 0; dy < 4; ++dy)
    for (int dx = 0; dx < 4; ++dx)
      s4 += xb[(yy * 4 + dy) * HW + (xx * 4 + dx)];
  int by = (yy >> 1) * 8, bx = (xx >> 1) * 8;
  float s8 = 0.f;
  for (int dy = 0; dy < 8; ++dy)
    for (int dx = 0; dx < 8; ++dx)
      s8 += xb[(by + dy) * HW + (bx + dx)];
  np[idx] = (_Float16)(s4 * (1.0f / 16.0f) - s8 * (1.0f / 64.0f));
}

__global__ void k_pos_mean(const float* __restrict__ pe, float* __restrict__ pm) {
  int d = blockIdx.x * blockDim.x + threadIdx.x;
  if (d >= DDIM) return;
  float s = 0.f;
  for (int t = 0; t < NTOK; ++t) s += pe[(size_t)t * DDIM + d];
  pm[d] = s * (1.0f / 256.0f);
}

// Weight K x N (f32, row-major) -> f16 transposed Bt[n][k], K zero-padded to Kp
__global__ void k_convert_wT(const float* __restrict__ W, _Float16* __restrict__ Wt,
                             int K, int N, int Kp) {
  int idx = blockIdx.x * blockDim.x + threadIdx.x;
  if (idx >= N * Kp) return;
  int n = idx / Kp, k = idx % Kp;
  Wt[idx] = (k < K) ? (_Float16)W[(size_t)k * N + n] : (_Float16)0.f;
}

// DCT^T directly in Bt form: Bt[kq][n] = s(kq) * cos(pi*(n+0.5)*kq/1024)
__global__ void k_dct_t16(_Float16* __restrict__ Bt) {
  int idx = blockIdx.x * blockDim.x + threadIdx.x;
  if (idx >= DDIM * DDIM) return;
  int kq = idx >> 10, n = idx & 1023;
  float s = (kq == 0) ? 0.03125f : 0.04419417382415922f; // sqrt(1/1024), sqrt(2/1024)
  const float PI = 3.14159265358979323846f;
  Bt[idx] = (_Float16)(s * __cosf(PI * ((float)n + 0.5f) * (float)kq * (1.0f / 1024.0f)));
}

// ---------------------------------------------------------------------------
// WMMA GEMM (guard-free): C[MxN] = A[MxKp] @ Bt[NxKp]^T (+bias)
// A, Bt: f16 row-major, lda/ldbt multiples of 8 halves (16B rows), Kp % 32 == 0,
// M % 64 == 0, N % 64 == 0. 128 threads = 4 waves; 64x64 block tile, BK=32.
// ---------------------------------------------------------------------------
#define BM 64
#define BN 64
#define BK 32
#define LDF 40  // LDS row stride in halves (80B -> 16B-aligned 8-half chunks)

__device__ __forceinline__ f16x16 load_frag(const _Float16* rowp, int kb) {
  // ISA 16-bit fragment: lanes 0-15 K {kb..kb+7, 16+kb..16+kb+7}, kb = 0 or 8
  f16x8 lo = *reinterpret_cast<const f16x8*>(rowp + kb);
  f16x8 hi = *reinterpret_cast<const f16x8*>(rowp + 16 + kb);
  return __builtin_shufflevector(lo, hi, 0, 1, 2, 3, 4, 5, 6, 7,
                                 8, 9, 10, 11, 12, 13, 14, 15);
}

__global__ __launch_bounds__(128)
void k_gemm_wmma(const _Float16* __restrict__ A, const _Float16* __restrict__ Bt,
                 float* __restrict__ C, _Float16* __restrict__ C16,
                 int M, int N, int Kp, int lda, int ldbt, int ldc,
                 const float* __restrict__ bias) {
  __shared__ _Float16 As[BM][LDF];
  __shared__ _Float16 Bs[BN][LDF];

  const int tid  = threadIdx.x;
  const int wave = tid >> 5;
  const int lane = tid & 31;
  const int n0   = blockIdx.x * BN;
  const int m0   = blockIdx.y * BM;

  f32x8 acc[4];
#pragma unroll
  for (int t = 0; t < 4; ++t)
#pragma unroll
    for (int e = 0; e < 8; ++e) acc[t][e] = 0.0f;

  // copy mapping: 64 rows x 4 chunks(8 halves); thread covers 2 chunks/matrix
  const int crow = tid & 63;
  const int seg0 = (tid >> 6) * 2;

  const int nsteps = Kp / BK;
  for (int ks = 0; ks < nsteps; ++ks) {
    const int k0 = ks * BK;
    const _Float16* ga = A  + (size_t)(m0 + crow) * lda  + k0;
    const _Float16* gb = Bt + (size_t)(n0 + crow) * ldbt + k0;
#pragma unroll
    for (int s = 0; s < 2; ++s) {
      copy16B_to_lds(ga + (seg0 + s) * 8, &As[crow][(seg0 + s) * 8]);
      copy16B_to_lds(gb + (seg0 + s) * 8, &Bs[crow][(seg0 + s) * 8]);
    }
    if (ks + 1 < nsteps) {           // speculative prefetch of next K tiles
      __builtin_prefetch(ga + BK, 0, 1);
      __builtin_prefetch(gb + BK, 0, 1);
    }
    wait_lds_copies();
    __syncthreads();

    const int frow = wave * 16 + (lane & 15);
    const int kb   = (lane < 16) ? 0 : 8;
    f16x16 afrag = load_frag(&As[frow][0], kb);
#pragma unroll
    for (int t = 0; t < 4; ++t) {
      f16x16 bfrag = load_frag(&Bs[t * 16 + (lane & 15)][0], kb);
      acc[t] = __builtin_amdgcn_wmma_f32_16x16x32_f16(
          false, afrag, false, bfrag, (short)0, acc[t], false, false);
    }
    __syncthreads();
  }

  // C/D layout: VGPR r holds M=r (lanes 0-15) / M=r+8 (lanes 16-31)
  const int mb = m0 + wave * 16 + ((lane < 16) ? 0 : 8);
#pragma unroll
  for (int t = 0; t < 4; ++t) {
    int n = n0 + t * 16 + (lane & 15);
    float bv = bias ? bias[n] : 0.0f;
#pragma unroll
    for (int r = 0; r < 8; ++r) {
      float val = acc[t][r] + bv;
      if (C)   C[(size_t)(mb + r) * ldc + n] = val;
      if (C16) C16[(size_t)(mb + r) * ldc + n] = (_Float16)val;
    }
  }
}

// ---------------------------------------------------------------------------
// Row layernorm: f32 in -> f16 out (feeds the next GEMM). Block = 1 row.
// ---------------------------------------------------------------------------
__global__ void k_layernorm(const float* __restrict__ in, _Float16* __restrict__ out,
                            const float* __restrict__ g, const float* __restrict__ bv,
                            long in_stride, long out_stride) {
  const float* xr = in + (long)blockIdx.x * in_stride;
  _Float16* yr = out + (long)blockIdx.x * out_stride;
  __shared__ float red[256];
  __shared__ float s_mu, s_rs;
  int tid = threadIdx.x;

  float s = 0.f;
  for (int d = tid; d < DDIM; d += 256) s += xr[d];
  red[tid] = s; __syncthreads();
  for (int st = 128; st > 0; st >>= 1) {
    if (tid < st) red[tid] += red[tid + st];
    __syncthreads();
  }
  if (tid == 0) s_mu = red[0] * (1.0f / DDIM);
  __syncthreads();
  float mu = s_mu;

  float v = 0.f;
  for (int d = tid; d < DDIM; d += 256) { float t = xr[d] - mu; v += t * t; }
  red[tid] = v; __syncthreads();
  for (int st = 128; st > 0; st >>= 1) {
    if (tid < st) red[tid] += red[tid + st];
    __syncthreads();
  }
  if (tid == 0) s_rs = rsqrtf(red[0] * (1.0f / DDIM) + 1e-5f);
  __syncthreads();
  float rs = s_rs;

  for (int d = tid; d < DDIM; d += 256)
    yr[d] = (_Float16)((xr[d] - mu) * rs * g[d] + bv[d]);
}

// ---------------------------------------------------------------------------
// 3-token attention: scores = q k^T / 32, softmax, o = a @ v (f16 out)
// ---------------------------------------------------------------------------
__global__ void k_attention(const float* __restrict__ q, const float* __restrict__ k,
                            const float* __restrict__ v, _Float16* __restrict__ o) {
  int b = blockIdx.x, tid = threadIdx.x;
  __shared__ float red[256];
  __shared__ float aw[9];
  const float* qb = q + (size_t)b * 3 * DDIM;
  const float* kb = k + (size_t)b * 3 * DDIM;
  const float* vb = v + (size_t)b * 3 * DDIM;

  float part[9];
#pragma unroll
  for (int p = 0; p < 9; ++p) part[p] = 0.f;
  for (int d = tid; d < DDIM; d += 256) {
    float q0 = qb[d], q1 = qb[DDIM + d], q2 = qb[2 * DDIM + d];
    float c0 = kb[d], c1 = kb[DDIM + d], c2 = kb[2 * DDIM + d];
    part[0] += q0 * c0; part[1] += q0 * c1; part[2] += q0 * c2;
    part[3] += q1 * c0; part[4] += q1 * c1; part[5] += q1 * c2;
    part[6] += q2 * c0; part[7] += q2 * c1; part[8] += q2 * c2;
  }
  for (int p = 0; p < 9; ++p) {
    red[tid] = part[p]; __syncthreads();
    for (int st = 128; st > 0; st >>= 1) {
      if (tid < st) red[tid] += red[tid + st];
      __syncthreads();
    }
    if (tid == 0) aw[p] = red[0] * (1.0f / 32.0f);  // 1/sqrt(1024)
    __syncthreads();
  }
  if (tid == 0) {
    for (int n = 0; n < 3; ++n) {
      float a0 = aw[3 * n], a1 = aw[3 * n + 1], a2 = aw[3 * n + 2];
      float mx = fmaxf(a0, fmaxf(a1, a2));
      float e0 = __expf(a0 - mx), e1 = __expf(a1 - mx), e2 = __expf(a2 - mx);
      float inv = 1.0f / (e0 + e1 + e2);
      aw[3 * n] = e0 * inv; aw[3 * n + 1] = e1 * inv; aw[3 * n + 2] = e2 * inv;
    }
  }
  __syncthreads();
  _Float16* ob = o + (size_t)b * 3 * DDIM;
  for (int d = tid; d < DDIM; d += 256) {
    float v0 = vb[d], v1 = vb[DDIM + d], v2 = vb[2 * DDIM + d];
#pragma unroll
    for (int n = 0; n < 3; ++n)
      ob[n * DDIM + d] =
          (_Float16)(aw[3 * n] * v0 + aw[3 * n + 1] * v1 + aw[3 * n + 2] * v2);
  }
}

// out[b] = sum_d mean_n(oproj[b,n,d] + tokens[b,n,d]) * W_fc[d] + b_fc
__global__ void k_final(const float* __restrict__ op, const float* __restrict__ tok,
                        const float* __restrict__ wfc, const float* __restrict__ bfc,
                        float* __restrict__ out) {
  int b = blockIdx.x, tid = threadIdx.x;
  __shared__ float red[256];
  float acc = 0.f;
  const float* ob = op  + (size_t)b * 3 * DDIM;
  const float* tb = tok + (size_t)b * 3 * DDIM;
  for (int d = tid; d < DDIM; d += 256) {
    float s = 0.f;
#pragma unroll
    for (int n = 0; n < 3; ++n) s += ob[n * DDIM + d] + tb[n * DDIM + d];
    acc += (s * (1.0f / 3.0f)) * wfc[d];
  }
  red[tid] = acc; __syncthreads();
  for (int st = 128; st > 0; st >>= 1) {
    if (tid < st) red[tid] += red[tid + st];
    __syncthreads();
  }
  if (tid == 0) out[b] = red[0] + bfc[0];
}

// ---------------------------------------------------------------------------
// Launch
// ---------------------------------------------------------------------------
extern "C" void kernel_launch(void* const* d_in, const int* in_sizes, int n_in,
                              void* d_out, int out_size, void* d_ws, size_t ws_size,
                              hipStream_t stream) {
  (void)in_sizes; (void)n_in; (void)out_size; (void)ws_size;

  const float* x       = (const float*)d_in[0];
  const int*   sidx    = (const int*)  d_in[1];
  const int*   rotk    = (const int*)  d_in[2];
  const int*   fliph   = (const int*)  d_in[3];
  const int*   flipw   = (const int*)  d_in[4];
  const float* W_pe    = (const float*)d_in[5];
  const float* pos_emb = (const float*)d_in[6];
  const float* ln_g    = (const float*)d_in[7];
  const float* ln_b    = (const float*)d_in[8];
  const float* W_noise = (const float*)d_in[9];
  const float* Wq      = (const float*)d_in[10];
  const float* Wk      = (const float*)d_in[11];
  const float* Wv      = (const float*)d_in[12];
  const float* Wo      = (const float*)d_in[13];
  const float* ah_g    = (const float*)d_in[14];
  const float* ah_b    = (const float*)d_in[15];
  const float* W_fc    = (const float*)d_in[16];
  const float* b_fc    = (const float*)d_in[17];
  float* out = (float*)d_out;

  // byte-based workspace allocator (256B aligned slots)
  char* wsb = (char*)d_ws;
  size_t off = 0;
  auto alloc = [&](size_t bytes) -> void* {
    void* p = wsb + off;
    off += (bytes + 255) & ~(size_t)255;
    return p;
  };
  _Float16* rA_shuf = (_Float16*)alloc((size_t)BATCH * PKPE * 2);
  _Float16* rA_orig = (_Float16*)alloc((size_t)BATCH * PKPE * 2);
  _Float16* npoolA  = (_Float16*)alloc((size_t)BATCH * NOISED * 2);
  float*    pos_m   = (float*)   alloc(DDIM * 4);
  _Float16* WpeT    = (_Float16*)alloc((size_t)DDIM * PKPE * 2);
  _Float16* WnoiseT = (_Float16*)alloc((size_t)DDIM * NOISED * 2);
  _Float16* WqT     = (_Float16*)alloc((size_t)DDIM * DDIM * 2);
  _Float16* WkT     = (_Float16*)alloc((size_t)DDIM * DDIM * 2);
  _Float16* WvT     = (_Float16*)alloc((size_t)DDIM * DDIM * 2);
  _Float16* WoT     = (_Float16*)alloc((size_t)DDIM * DDIM * 2);
  _Float16* dctT    = (_Float16*)alloc((size_t)DDIM * DDIM * 2);
  float*    t_shuf  = (float*)   alloc((size_t)BATCH * DDIM * 4);
  float*    t_orig  = (float*)   alloc((size_t)BATCH * DDIM * 4);
  _Float16* featsA  = (_Float16*)alloc((size_t)BATCH * 3 * DDIM * 2);
  float*    dctb    = (float*)   alloc((size_t)BATCH * 3 * DDIM * 4);
  _Float16* tln     = (_Float16*)alloc((size_t)BATCH * 3 * DDIM * 2);
  float*    qb      = (float*)   alloc((size_t)BATCH * 3 * DDIM * 4);
  float*    kbuf    = (float*)   alloc((size_t)BATCH * 3 * DDIM * 4);
  float*    vb      = (float*)   alloc((size_t)BATCH * 3 * DDIM * 4);
  _Float16* obA     = (_Float16*)alloc((size_t)BATCH * 3 * DDIM * 2);
  float*    op      = (float*)   alloc((size_t)BATCH * 3 * DDIM * 4);

  const int T = 256;
  // 1) activation reductions (f16, padded K) + bias + weight transforms
  k_token_mean     <<<(BATCH * PKPE   + T - 1) / T, T, 0, stream>>>(x, rA_orig);
  k_token_mean_shuf<<<(BATCH * PKPE   + T - 1) / T, T, 0, stream>>>(x, sidx, rotk, fliph, flipw, rA_shuf);
  k_npool          <<<(BATCH * NOISED + T - 1) / T, T, 0, stream>>>(x, npoolA);
  k_pos_mean       <<<(DDIM + T - 1) / T, T, 0, stream>>>(pos_emb, pos_m);
  k_convert_wT<<<((size_t)DDIM * PKPE   + T - 1) / T, T, 0, stream>>>(W_pe,    WpeT,    PDIM,   DDIM, PKPE);
  k_convert_wT<<<((size_t)DDIM * NOISED + T - 1) / T, T, 0, stream>>>(W_noise, WnoiseT, NOISED, DDIM, NOISED);
  k_convert_wT<<<((size_t)DDIM * DDIM   + T - 1) / T, T, 0, stream>>>(Wq, WqT, DDIM, DDIM, DDIM);
  k_convert_wT<<<((size_t)DDIM * DDIM   + T - 1) / T, T, 0, stream>>>(Wk, WkT, DDIM, DDIM, DDIM);
  k_convert_wT<<<((size_t)DDIM * DDIM   + T - 1) / T, T, 0, stream>>>(Wv, WvT, DDIM, DDIM, DDIM);
  k_convert_wT<<<((size_t)DDIM * DDIM   + T - 1) / T, T, 0, stream>>>(Wo, WoT, DDIM, DDIM, DDIM);
  k_dct_t16   <<<((size_t)DDIM * DDIM   + T - 1) / T, T, 0, stream>>>(dctT);

  // 2) branch GEMMs
  dim3 g64(DDIM / BN, 1);
  dim3 g192(DDIM / BN, (3 * BATCH) / BM);
  k_gemm_wmma<<<g64, 128, 0, stream>>>(rA_shuf, WpeT, t_shuf, nullptr,
                                       BATCH, DDIM, PKPE, PKPE, PKPE, DDIM, pos_m);
  k_gemm_wmma<<<g64, 128, 0, stream>>>(rA_orig, WpeT, t_orig, nullptr,
                                       BATCH, DDIM, PKPE, PKPE, PKPE, DDIM, pos_m);
  k_gemm_wmma<<<g64, 128, 0, stream>>>(npoolA, WnoiseT, nullptr, featsA + 2 * DDIM,
                                       BATCH, DDIM, NOISED, NOISED, NOISED, 3 * DDIM, nullptr);

  // 3) ln_post on shuffle/orig branches -> f16 rows of feats (b*3 + branch)
  k_layernorm<<<BATCH, T, 0, stream>>>(t_shuf, featsA + 0 * DDIM, ln_g, ln_b, DDIM, 3 * DDIM);
  k_layernorm<<<BATCH, T, 0, stream>>>(t_orig, featsA + 1 * DDIM, ln_g, ln_b, DDIM, 3 * DDIM);

  // 4) DCT: feats(192x1024) @ dctT^T
  k_gemm_wmma<<<g192, 128, 0, stream>>>(featsA, dctT, dctb, nullptr,
                                        3 * BATCH, DDIM, DDIM, DDIM, DDIM, DDIM, nullptr);

  // 5) attention head
  k_layernorm<<<3 * BATCH, T, 0, stream>>>(dctb, tln, ah_g, ah_b, DDIM, DDIM);
  k_gemm_wmma<<<g192, 128, 0, stream>>>(tln, WqT, qb, nullptr,
                                        3 * BATCH, DDIM, DDIM, DDIM, DDIM, DDIM, nullptr);
  k_gemm_wmma<<<g192, 128, 0, stream>>>(tln, WkT, kbuf, nullptr,
                                        3 * BATCH, DDIM, DDIM, DDIM, DDIM, DDIM, nullptr);
  k_gemm_wmma<<<g192, 128, 0, stream>>>(tln, WvT, vb, nullptr,
                                        3 * BATCH, DDIM, DDIM, DDIM, DDIM, DDIM, nullptr);
  k_attention<<<BATCH, T, 0, stream>>>(qb, kbuf, vb, obA);
  k_gemm_wmma<<<g192, 128, 0, stream>>>(obA, WoT, op, nullptr,
                                        3 * BATCH, DDIM, DDIM, DDIM, DDIM, DDIM, nullptr);

  // 6) residual + token mean + fc
  k_final<<<BATCH, T, 0, stream>>>(op, dctb, W_fc, b_fc, out);
}
